// VQVAE_64063732187190
// MI455X (gfx1250) — compile-verified, matrix operand order checked by reading, not run on hardware
//
#include <hip/hip_runtime.h>
#include <hip/hip_bf16.h>
#include <stdint.h>

// ---------------------------------------------------------------------------
// CDNA5 (gfx1250) wave32 WMMA types
// ---------------------------------------------------------------------------
typedef __attribute__((ext_vector_type(16))) __bf16 v16bf;
typedef __attribute__((ext_vector_type(8)))  __bf16 v8bf;
typedef __attribute__((ext_vector_type(2)))  __bf16 v2bf;
typedef __attribute__((ext_vector_type(8)))  float  v8f;

// ---------------------------------------------------------------------------
// One 16x16x32 K-chunk of the implicit-GEMM conv:
//  - stage 16x32 weight tile to LDS (1 packed bf16x2 store per thread)
//  - A fragment = two contiguous ds_load_b128 per lane (CDNA5 16-bit A layout:
//    lane row = lane%16, K runs [kb..kb+7] and [16+kb..16+kb+7], kb=8*(lane/16))
//  - B fragment = branch-free predicated im2col gather (offset clamped to 0,
//    result selected) so the 16 loads pipeline without exec-mask branches
// ---------------------------------------------------------------------------
template <int KW, int STRIDE, int MODE, bool TAIL>
__device__ __forceinline__ void conv_chunk(
    const float* __restrict__ xb,   // x + b*Ci*Tin (always a valid pointer)
    const float* __restrict__ w,
    __bf16* wlds, int tid, int lane, int kb,
    int coBase, int Co, int KK, int kc,
    int Tin, int tS_p, bool colOK, v8f& acc)
{
  // ---- stage weight pair (row m, cols k0,k0+1) ----
  {
    const int m  = tid >> 4;
    const int k0 = (tid & 15) << 1;
    const int co = coBase + m;
    const bool rowOK = (co < Co);
    const int kk0 = kc + k0;
    const bool ok0 = rowOK && (!TAIL || kk0 < KK);
    const bool ok1 = rowOK && (!TAIL || kk0 + 1 < KK);
    size_t off0, off1;
    if (MODE == 0) {
      off0 = (size_t)co * KK + kk0;
      off1 = off0 + 1;
    } else {
      const int ci0 = kk0 / KW, j0 = kk0 - ci0 * KW;        // constant divisor
      const int kk1 = kk0 + 1;
      const int ci1 = kk1 / KW, j1 = kk1 - ci1 * KW;
      off0 = (size_t)co * KK + ci0 * KW + (KW - 1 - j0);
      off1 = (size_t)co * KK + ci1 * KW + (KW - 1 - j1);
    }
    const float t0 = w[ok0 ? off0 : 0];
    const float t1 = w[ok1 ? off1 : 0];
    v2bf p;
    p[0] = (__bf16)(ok0 ? t0 : 0.f);
    p[1] = (__bf16)(ok1 ? t1 : 0.f);
    *reinterpret_cast<v2bf*>(&wlds[m * 32 + k0]) = p;       // one ds_store_b32
  }
  __syncthreads();

  // ---- A fragment: two 16B LDS vector loads ----
  const int row = lane & 15;
  const v8bf lo = *reinterpret_cast<const v8bf*>(&wlds[row * 32 + kb]);
  const v8bf hi = *reinterpret_cast<const v8bf*>(&wlds[row * 32 + 16 + kb]);
  const v16bf au = __builtin_shufflevector(lo, hi,
      0, 1, 2, 3, 4, 5, 6, 7, 8, 9, 10, 11, 12, 13, 14, 15);

  // ---- B fragment: predicated im2col gather ----
  v16bf bu;
#pragma unroll
  for (int i = 0; i < 16; ++i) {
    const int kl = (i < 8 ? i : i + 8) + kb;  // [kb..kb+7], [16+kb..16+kb+7]
    const int kk = kc + kl;
    const int ci = kk / KW, j = kk - ci * KW;               // constant divisor
    bool ok = colOK && (!TAIL || kk < KK);
    size_t off = 0;
    if (MODE == 0) {
      const int u = tS_p + j;
      ok = ok && (u >= 0) && (u < Tin);
      off = (size_t)ci * Tin + (size_t)(u >= 0 ? u : 0);
    } else {
      const int u = tS_p + j;
      const int ui = u / STRIDE;                            // STRIDE==2: shift
      ok = ok && (u >= 0) && ((u % STRIDE) == 0) && (ui < Tin);
      off = (size_t)ci * Tin + (size_t)(ui >= 0 ? ui : 0);
    }
    const float tv = xb[ok ? off : 0];
    bu[i] = (__bf16)(ok ? tv : 0.f);
  }

  acc = __builtin_amdgcn_wmma_f32_16x16x32_bf16(
      false, au, false, bu, (short)0, acc, false, false);
  __syncthreads();
}

// ---------------------------------------------------------------------------
// Generic implicit-GEMM conv1d / convT1d on the WMMA pipe.
//   MODE 0 (conv):    y[b,co,t] = bias[co] + sum_{ci,j} w[co,ci,j]      * x[b,ci,t*S-p+j]
//   MODE 1 (deconv):  y[b,co,t] = bias[co] + sum_{ci,j} w[co,ci,KW-1-j] * x[b,ci,(t-(KW-1-p)+j)/S]
// Block = 256 threads (8 waves). Block tile: 16 (co) x 128 (n = b*Tout+t).
//   act 0: none, 2: tanh
// ---------------------------------------------------------------------------
template <int KW, int STRIDE, int MODE>
__global__ __launch_bounds__(256)
void conv1d_wmma(const float* __restrict__ x, const float* __restrict__ w,
                 const float* __restrict__ bias, float* __restrict__ y,
                 int B, int Ci, int Co, int Tin, int Tout, int pad, int act)
{
  __shared__ __align__(16) __bf16 wlds[16 * 32];
  const int tid    = threadIdx.x;
  const int lane   = tid & 31;
  const int wave   = tid >> 5;
  const int coBase = blockIdx.y * 16;
  const int nBase  = blockIdx.x * 128 + wave * 16;
  const int Ntot   = B * Tout;
  const int KK     = Ci * KW;
  const int kb     = (lane >> 4) * 8;

  const int nCol = nBase + (lane & 15);
  int b = 0, t = 0;
  const bool colOK = (nCol < Ntot);
  if (colOK) { b = nCol / Tout; t = nCol - b * Tout; }
  const float* xb = x + (size_t)b * Ci * Tin;
  const int tS_p = (MODE == 0) ? (t * STRIDE - pad) : (t - (KW - 1 - pad));

  v8f acc = {};
  const int nFull = KK >> 5;
  for (int c = 0; c < nFull; ++c) {
    if (tid == 0 && c + 1 < nFull)   // pull next weight chunk toward L2
      __builtin_prefetch(&w[(size_t)coBase * KK + (c << 5) + 32], 0, 1);
    conv_chunk<KW, STRIDE, MODE, false>(xb, w, wlds, tid, lane, kb,
                                        coBase, Co, KK, c << 5, Tin, tS_p, colOK, acc);
  }
  if (KK & 31)
    conv_chunk<KW, STRIDE, MODE, true>(xb, w, wlds, tid, lane, kb,
                                       coBase, Co, KK, nFull << 5, Tin, tS_p, colOK, acc);

  // ---- epilogue: bias (+ optional tanh), f32 C/D layout store ----
#pragma unroll
  for (int r = 0; r < 8; ++r) {
    const int m  = r + ((lane >> 4) << 3);
    const int co = coBase + m;
    if (co < Co && colOK) {
      float v = acc[r] + (bias ? bias[co] : 0.f);
      if (act == 2) v = tanhf(v);
      y[((size_t)b * Co + co) * Tout + t] = v;
    }
  }
}

// ---------------------------------------------------------------------------
// BatchNorm1d (training mode, gamma=1/beta=0): per-channel mean / invstd
// ---------------------------------------------------------------------------
__global__ __launch_bounds__(256)
void bn_stats(const float* __restrict__ x, float* __restrict__ mean,
              float* __restrict__ invstd, int B, int C, int T)
{
  __shared__ float ssum[256], ssq[256];
  const int c = blockIdx.x;
  const int n = B * T;
  float s = 0.f, q = 0.f;
  for (int i = threadIdx.x; i < n; i += 256) {
    const int b = i / T, t = i - b * T;
    const float v = x[((size_t)b * C + c) * T + t];
    s += v; q += v * v;
  }
  ssum[threadIdx.x] = s; ssq[threadIdx.x] = q;
  __syncthreads();
  for (int o = 128; o > 0; o >>= 1) {
    if (threadIdx.x < o) {
      ssum[threadIdx.x] += ssum[threadIdx.x + o];
      ssq[threadIdx.x]  += ssq[threadIdx.x + o];
    }
    __syncthreads();
  }
  if (threadIdx.x == 0) {
    const float m   = ssum[0] / (float)n;
    const float var = ssq[0] / (float)n - m * m;
    mean[c]   = m;
    invstd[c] = rsqrtf(var + 1e-5f);
  }
}

__global__ void bn_relu(float* __restrict__ y, const float* __restrict__ mean,
                        const float* __restrict__ invstd, int C, int T, int total)
{
  const int i = blockIdx.x * blockDim.x + threadIdx.x;
  if (i >= total) return;
  const int c = (i / T) % C;
  const float v = (y[i] - mean[c]) * invstd[c];
  y[i] = v > 0.f ? v : 0.f;
}

// ---------------------------------------------------------------------------
// VQ: codebook squared norms, argmin over (||cb||^2 - 2*S), gather
// S is produced by conv1d_wmma reuse and stored as S[k*1024 + n].
// ---------------------------------------------------------------------------
__global__ __launch_bounds__(256)
void code_norms(const float* __restrict__ cb, float* __restrict__ cbn)
{
  __shared__ float s[256];
  const int k = blockIdx.x;
  const float v = cb[(size_t)k * 256 + threadIdx.x];
  s[threadIdx.x] = v * v;
  __syncthreads();
  for (int o = 128; o > 0; o >>= 1) {
    if (threadIdx.x < o) s[threadIdx.x] += s[threadIdx.x + o];
    __syncthreads();
  }
  if (threadIdx.x == 0) cbn[k] = s[0];
}

__global__ __launch_bounds__(256)
void vq_argmin(const float* __restrict__ S, const float* __restrict__ cbn,
               int* __restrict__ idx)
{
  __shared__ float sd[256];
  __shared__ int   si[256];
  const int n = blockIdx.x;
  float best = 3.4e38f; int bi = 0x7FFFFFFF;
  for (int k = threadIdx.x; k < 1024; k += 256) {
    const float d = cbn[k] - 2.f * S[(size_t)k * 1024 + n];
    if (d < best || (d == best && k < bi)) { best = d; bi = k; }
  }
  sd[threadIdx.x] = best; si[threadIdx.x] = bi;
  __syncthreads();
  for (int o = 128; o > 0; o >>= 1) {
    if (threadIdx.x < o) {
      const float d2 = sd[threadIdx.x + o]; const int i2 = si[threadIdx.x + o];
      if (d2 < sd[threadIdx.x] || (d2 == sd[threadIdx.x] && i2 < si[threadIdx.x])) {
        sd[threadIdx.x] = d2; si[threadIdx.x] = i2;
      }
    }
    __syncthreads();
  }
  if (threadIdx.x == 0) idx[n] = si[0];
}

// zf is just z viewed flat: zf[n,d] = zflat[n*256+d]; forward value is exactly zq.
__global__ void vq_gather(const float* __restrict__ cb, const int* __restrict__ idx,
                          float* __restrict__ y, int total)
{
  const int i = blockIdx.x * blockDim.x + threadIdx.x;
  if (i >= total) return;
  y[i] = cb[(size_t)idx[i >> 8] * 256 + (i & 255)];
}

// ---------------------------------------------------------------------------
// Host-side dispatch over the template instantiations
// ---------------------------------------------------------------------------
static inline void launch_conv(const float* x, const float* w, const float* bias,
                               float* y, int B, int Ci, int Co, int Tin, int Tout,
                               int Kw, int pad, int mode, int act, hipStream_t s)
{
  const int N = B * Tout;
  dim3 grid((N + 127) / 128, (Co + 15) / 16);
  if (mode == 0) {
    switch (Kw) {
      case 1:  conv1d_wmma<1, 1, 0><<<grid, 256, 0, s>>>(x, w, bias, y, B, Ci, Co, Tin, Tout, pad, act); break;
      case 2:  conv1d_wmma<2, 2, 0><<<grid, 256, 0, s>>>(x, w, bias, y, B, Ci, Co, Tin, Tout, pad, act); break;
      case 4:  conv1d_wmma<4, 2, 0><<<grid, 256, 0, s>>>(x, w, bias, y, B, Ci, Co, Tin, Tout, pad, act); break;
      case 5:  conv1d_wmma<5, 2, 0><<<grid, 256, 0, s>>>(x, w, bias, y, B, Ci, Co, Tin, Tout, pad, act); break;
      default: conv1d_wmma<11, 2, 0><<<grid, 256, 0, s>>>(x, w, bias, y, B, Ci, Co, Tin, Tout, pad, act); break;
    }
  } else {
    switch (Kw) {
      case 2:  conv1d_wmma<2, 2, 1><<<grid, 256, 0, s>>>(x, w, bias, y, B, Ci, Co, Tin, Tout, pad, act); break;
      case 4:  conv1d_wmma<4, 2, 1><<<grid, 256, 0, s>>>(x, w, bias, y, B, Ci, Co, Tin, Tout, pad, act); break;
      case 5:  conv1d_wmma<5, 2, 1><<<grid, 256, 0, s>>>(x, w, bias, y, B, Ci, Co, Tin, Tout, pad, act); break;
      default: conv1d_wmma<11, 2, 1><<<grid, 256, 0, s>>>(x, w, bias, y, B, Ci, Co, Tin, Tout, pad, act); break;
    }
  }
}

// ---------------------------------------------------------------------------
// Host orchestration
// ---------------------------------------------------------------------------
extern "C" void kernel_launch(void* const* d_in, const int* in_sizes, int n_in,
                              void* d_out, int out_size, void* d_ws, size_t ws_size,
                              hipStream_t stream)
{
  (void)in_sizes; (void)n_in; (void)out_size; (void)ws_size;
  const float* x = (const float*)d_in[0];
  const float* enc_w[10], *enc_b[10], *dec_w[10], *dec_b[10];
  for (int i = 0; i < 10; ++i) {
    enc_w[i] = (const float*)d_in[1 + i];
    enc_b[i] = (const float*)d_in[11 + i];
    dec_w[i] = (const float*)d_in[21 + i];
    dec_b[i] = (const float*)d_in[31 + i];
  }
  const float* cb = (const float*)d_in[41];

  // workspace layout (needs ~140 MB): two 64MB activation ping-pong buffers,
  // 4MB VQ score matrix, small stats/index arrays.
  const size_t BUF = 16777216ull;                    // 64 * 32 * 8192 floats
  float* bufA  = (float*)d_ws;
  float* bufB  = bufA + BUF;
  float* Smat  = bufB + BUF;                          // 1024*1024 f32
  float* cbn   = Smat + 1024ull * 1024ull;            // 1024
  float* bmean = cbn + 1024;                          // up to 256
  float* binv  = bmean + 512;                         // up to 256
  int*   idx   = (int*)(binv + 512);                  // 1024

  static const int encT [11] = {16384,8192,4096,2048,1024,512,256,130,65,32,16};
  static const int encCi[10] = {1,32,64,64,128,128,128,256,256,256};
  static const int encCo[10] = {32,64,64,128,128,128,256,256,256,256};
  static const int encK [10] = {11,5,4,4,4,4,4,4,2,2};
  static const int encP [10] = {5,2,1,1,1,1,3,1,0,0};

  static const int decT [11] = {16,32,64,128,253,506,1012,2024,4048,8096,16192};
  static const int decCi[10] = {256,256,256,256,128,128,128,64,64,32};
  static const int decCo[10] = {256,256,256,128,128,128,64,64,32,1};
  static const int decK [10] = {2,2,4,4,4,4,4,4,5,11};
  static const int decP [10] = {0,0,1,3,1,1,1,1,2,5};

  const float* cur = x;
  float* nxt = bufA;

  // ---- Encoder ----
  for (int i = 0; i < 10; ++i) {
    const int Tin = encT[i], Tout = encT[i + 1];
    const int Ci = encCi[i], Co = encCo[i];
    launch_conv(cur, enc_w[i], enc_b[i], nxt, 64, Ci, Co, Tin, Tout,
                encK[i], encP[i], /*mode=*/0, /*act=*/0, stream);
    if (i < 9) {
      bn_stats<<<Co, 256, 0, stream>>>(nxt, bmean, binv, 64, Co, Tout);
      const int total = 64 * Co * Tout;
      bn_relu<<<(total + 255) / 256, 256, 0, stream>>>(nxt, bmean, binv, Co, Tout, total);
    }
    cur = nxt;
    nxt = (nxt == bufA) ? bufB : bufA;
  }

  // ---- VQ: S[n,k] = zf @ cb^T via the same WMMA kernel (1x1 "conv") ----
  code_norms<<<1024, 256, 0, stream>>>(cb, cbn);
  // A := z rows (M=1024, K=256) as "weights"; batch := 1024 codebook rows.
  launch_conv(cb, cur, nullptr, Smat, 1024, 256, 1024, 1, 1,
              /*Kw=*/1, /*pad=*/0, /*mode=*/0, /*act=*/0, stream);
  vq_argmin<<<1024, 256, 0, stream>>>(Smat, cbn, idx);
  vq_gather<<<(262144 + 255) / 256, 256, 0, stream>>>(cb, idx, nxt, 262144);
  cur = nxt;
  nxt = (nxt == bufA) ? bufB : bufA;

  // ---- Decoder (ConvTranspose1d == input-dilated conv with flipped taps) ----
  for (int i = 0; i < 10; ++i) {
    const int Tin = decT[i], Tout = decT[i + 1];
    const int Ci = decCi[i], Co = decCo[i];
    float* out = (i == 9) ? (float*)d_out : nxt;
    launch_conv(cur, dec_w[i], dec_b[i], out, 64, Ci, Co, Tin, Tout,
                decK[i], decP[i], /*mode=*/1, /*act=*/(i == 9) ? 2 : 0, stream);
    if (i < 9) {
      bn_stats<<<Co, 256, 0, stream>>>(out, bmean, binv, 64, Co, Tout);
      const int total = 64 * Co * Tout;
      bn_relu<<<(total + 255) / 256, 256, 0, stream>>>(out, bmean, binv, Co, Tout, total);
    }
    cur = out;
    nxt = (nxt == bufA) ? bufB : bufA;
  }
}
